// Classifier_17549236371650
// MI455X (gfx1250) — matrix operand order repaired and checked by reading, not compiled, not run on hardware
//
#include <hip/hip_runtime.h>
#include <hip/hip_bf16.h>
#include <math.h>

// ---------------------------------------------------------------------------
// Types for CDNA5 WMMA / TDM (gfx1250, wave32)
// ---------------------------------------------------------------------------
typedef __attribute__((ext_vector_type(16))) __bf16       v16bf;
typedef __attribute__((ext_vector_type(8)))  float        v8f;
typedef __attribute__((ext_vector_type(4)))  unsigned int u32x4;
typedef __attribute__((ext_vector_type(8)))  int          i32x8;
typedef __attribute__((ext_vector_type(4)))  int          i32x4;
typedef __attribute__((address_space(3)))    __bf16       lds_bf16;

#define C_CH   192
#define POOLSZ 7
#define NROIS  1000
#define MPAD   1024                       // NROIS padded to GEMM M-tile
#define KFC1   (C_CH * POOLSZ * POOLSZ)   // 9408
#define DHID   768

#if defined(__HIP_DEVICE_COMPILE__) && \
    __has_builtin(__builtin_amdgcn_tensor_load_to_lds) && \
    __has_builtin(__builtin_amdgcn_s_wait_tensorcnt)
#define USE_TDM 1
#else
#define USE_TDM 0
#endif

#if USE_TDM
// 2D bf16 tile load: tileK elems (must be 32 -> 64B rows) x tileRows rows,
// global row stride = strideElems, into LDS at ldsByteAddr with an extra
// 16B pad after every 64B row (=> 40-element LDS row stride).
__device__ __forceinline__ void tdm_load_tile_bf16(unsigned int ldsByteAddr,
                                                   const __bf16* gptr,
                                                   unsigned int tileK,
                                                   unsigned int tileRows,
                                                   unsigned long long strideElems)
{
    const unsigned long long ga = (unsigned long long)(uintptr_t)gptr;
    u32x4 g0;
    g0[0] = 1u;                                        // count=1, no gather
    g0[1] = ldsByteAddr;                               // lds_addr
    g0[2] = (unsigned int)(ga & 0xFFFFFFFFull);        // global_addr[31:0]
    g0[3] = (unsigned int)((ga >> 32) & 0x01FFFFFFu)   // global_addr[56:32]
          | (2u << 30);                                // type = 2 (image)
    const unsigned long long td0 = 0x7FFFFFFFull;      // huge: no OOB clipping
    const unsigned long long td1 = 0x7FFFFFFFull;
    i32x8 g1;
    g1[0] = (int)((1u << 16)          // data_size = 2 bytes (bf16)
                | (1u << 20)          // pad_enable
                | (3u << 22)          // pad_interval code 3 = 16 DWORDs (64B)
                | (3u << 25));        // pad_amount  code 3 = 4 DWORDs (16B)
    g1[1] = (int)((td0 & 0xFFFFull) << 16);                          // abar=0 | td0.lo
    g1[2] = (int)(((td0 >> 16) & 0xFFFFull) | ((td1 & 0xFFFFull) << 16));
    g1[3] = (int)(((td1 >> 16) & 0xFFFFull) | (tileK << 16));        // tile_dim0
    g1[4] = (int)(tileRows & 0xFFFFu);                               // tile_dim1
    g1[5] = (int)(strideElems & 0xFFFFFFFFull);                      // dim0_stride lo
    g1[6] = (int)((strideElems >> 32) & 0xFFFFull);                  // dim0_stride hi
    g1[7] = 0;
    const i32x4 z4 = {0, 0, 0, 0};                     // groups 2/3 unused (2D)
    const i32x8 z8 = {0, 0, 0, 0, 0, 0, 0, 0};         // trailing group unused
    __builtin_amdgcn_tensor_load_to_lds(g0, g1, z4, z4, z8, 0);
}
#endif

// ---------------------------------------------------------------------------
// Kernel 0: fp32 -> bf16 weight conversion (4 elems / thread)
// ---------------------------------------------------------------------------
__global__ __launch_bounds__(256)
void cvt_f32_bf16_x4(const float4* __restrict__ s, __bf16* __restrict__ d, int n4)
{
    const int i = blockIdx.x * 256 + threadIdx.x;
    if (i < n4) {
        const float4 v = s[i];
        union { __bf16 h[4]; uint2 u; } p;
        p.h[0] = (__bf16)v.x; p.h[1] = (__bf16)v.y;
        p.h[2] = (__bf16)v.z; p.h[3] = (__bf16)v.w;
        *(uint2*)&d[(size_t)i * 4] = p.u;
    }
}

// ---------------------------------------------------------------------------
// Kernel 1: pyramid RoIAlign -> pooled bf16 [MPAD, 9408] (rows >= 1000 untouched)
// ---------------------------------------------------------------------------
__global__ __launch_bounds__(256)
void roi_align_kernel(const float* __restrict__ f0, const float* __restrict__ f1,
                      const float* __restrict__ f2, const float* __restrict__ f3,
                      const float* __restrict__ rois, __bf16* __restrict__ pooled)
{
    const int roi = blockIdx.x;
    const float b0 = rois[roi * 5 + 0];
    const float b1 = rois[roi * 5 + 1];
    const float b2 = rois[roi * 5 + 2];
    const float b3 = rois[roi * 5 + 3];
    const int   bix = (int)rois[roi * 5 + 4];

    // level = clip(round(4 + 0.5*log2(h*w)), 0, 3); rintf = round-half-even
    int lvl = (int)rintf(4.0f + 0.5f * log2f((b2 - b0) * (b3 - b1)));
    lvl = min(max(lvl, 0), 3);

    int H, W;
    const float* fbase;
    if (lvl == 0)      { H = 128; W = 128; fbase = f0; }
    else if (lvl == 1) { H = 64;  W = 64;  fbase = f1; }
    else if (lvl == 2) { H = 32;  W = 32;  fbase = f2; }
    else               { H = 16;  W = 16;  fbase = f3; }
    const int HW = H * W;
    fbase += (size_t)bix * C_CH * HW;

    const float y1 = b0 * (float)H, x1 = b1 * (float)W;
    const float y2 = b2 * (float)H, x2 = b3 * (float)W;
    const float bin_h = fmaxf(y2 - y1, 1.0f) / (float)POOLSZ;
    const float bin_w = fmaxf(x2 - x1, 1.0f) / (float)POOLSZ;

    __shared__ int   iy0[14], iy1[14], ix0[14], ix1[14];
    __shared__ float fly[14], flx[14], fvy[14], fvx[14];

    const int t = threadIdx.x;
    if (t < 28) {
        const int  k   = t % 14;
        const bool isY = (t < 14);
        const int p = k >> 1, s = k & 1;
        const float g = (float)p + ((float)s + 0.5f) * 0.5f;
        const float coord = isY ? (y1 + g * bin_h) : (x1 + g * bin_w);
        const int   lim   = isY ? H : W;
        const float v  = (coord >= -1.0f && coord <= (float)lim) ? 1.0f : 0.0f;
        const float cc = fminf(fmaxf(coord, 0.0f), (float)(lim - 1));
        const int c0 = (int)floorf(cc);
        const int c1 = min(c0 + 1, lim - 1);
        if (isY) { iy0[k] = c0; iy1[k] = c1; fly[k] = cc - (float)c0; fvy[k] = v; }
        else     { ix0[k] = c0; ix1[k] = c1; flx[k] = cc - (float)c0; fvx[k] = v; }
    }
    __syncthreads();

    __bf16* out = pooled + (size_t)roi * KFC1;
    for (int e = t; e < KFC1; e += 256) {
        const int c  = e / 49;
        const int r  = e % 49;
        const int py = r / 7, px = r % 7;
        const float* plane = fbase + (size_t)c * HW;
        float acc = 0.0f;
#pragma unroll
        for (int sy = 0; sy < 2; ++sy) {
            const int iy = py * 2 + sy;
            const int yA = iy0[iy], yB = iy1[iy];
            const float ly = fly[iy], vy = fvy[iy];
#pragma unroll
            for (int sx = 0; sx < 2; ++sx) {
                const int ix = px * 2 + sx;
                const int xA = ix0[ix], xB = ix1[ix];
                const float lx = flx[ix], vx = fvx[ix];
                const float f00 = plane[yA * W + xA];
                const float f01 = plane[yA * W + xB];
                const float f10 = plane[yB * W + xA];
                const float f11 = plane[yB * W + xB];
                const float val = f00 * (1.0f - ly) * (1.0f - lx)
                                + f01 * (1.0f - ly) * lx
                                + f10 * ly * (1.0f - lx)
                                + f11 * ly * lx;
                acc += val * vy * vx;
            }
        }
        out[e] = (__bf16)(acc * 0.25f);
    }
}

// ---------------------------------------------------------------------------
// Kernel 2: GEMM  out[M,N] = relu(A[M,K] @ Bt[N,K]^T + bias[N])
// A, Bt already bf16. Block tile 128(M) x 64(N), K-tile 32, DOUBLE-BUFFERED:
// wave 0 keeps the TDM pair for tile t+1 in flight while all 8 waves compute
// tile t; since TENSORcnt retires in order, s_wait_tensorcnt(2) releases the
// oldest (current) pair. 8 waves as 4(M) x 2(N); each wave 32x32 -> 4 v_wmma
// per K-step. LDS rows are 40 elements (64B data + 16B TDM pad) to stagger
// banks. A-buffers padded to MPAD rows -> unconditional loads; stores guarded.
// ---------------------------------------------------------------------------
__global__ __launch_bounds__(256)
void gemm_bf16_wmma(const __bf16* __restrict__ A, const __bf16* __restrict__ Bt,
                    const float* __restrict__ bias,
                    float* __restrict__ outF, __bf16* __restrict__ outB,
                    int M, int N, int K, int relu)
{
    __shared__ __align__(16) __bf16 As[2][128][40];
    __shared__ __align__(16) __bf16 Bs[2][64][40];

    const int tid   = threadIdx.x;
    const int mBase = blockIdx.y * 128;
    const int nBase = blockIdx.x * 64;
    const int w     = tid >> 5;
    const int lane  = tid & 31;
    const int m0    = (w & 3) * 32;   // wave M offset (two 16-row tiles)
    const int n0    = (w >> 2) * 32;  // wave N offset (two 16-col tiles)
    const int hi    = lane >> 4;
    const int lm    = lane & 15;

    const __bf16* gA = A  + (size_t)mBase * K;
    const __bf16* gB = Bt + (size_t)nBase * K;

#if USE_TDM
    const unsigned int asA0 = (unsigned int)(uintptr_t)(lds_bf16*)&As[0][0][0];
    const unsigned int asA1 = (unsigned int)(uintptr_t)(lds_bf16*)&As[1][0][0];
    const unsigned int bsA0 = (unsigned int)(uintptr_t)(lds_bf16*)&Bs[0][0][0];
    const unsigned int bsA1 = (unsigned int)(uintptr_t)(lds_bf16*)&Bs[1][0][0];
    // Prologue: fill buffer 0 with tile 0.
    if (w == 0) {
        tdm_load_tile_bf16(asA0, gA, 32u, 128u, (unsigned long long)K);
        tdm_load_tile_bf16(bsA0, gB, 32u,  64u, (unsigned long long)K);
    }
#else
    const int rA = tid >> 2;            // staging coords (4 x 16B chunks / row)
    const int cA = (tid & 3) * 8;
#endif

    v8f acc00 = {}, acc01 = {}, acc10 = {}, acc11 = {};

    const int nT = K >> 5;              // number of 32-wide K tiles
    for (int t = 0; t < nT; ++t) {
        const int p = t & 1;
#if USE_TDM
        if (w == 0) {
            if (t + 1 < nT) {
                // Prefetch tile t+1 into the other buffer, then retire only
                // the oldest pair (tile t): in-order TENSORcnt completion.
                const int ktn = (t + 1) << 5;
                tdm_load_tile_bf16(p ? asA0 : asA1, gA + ktn, 32u, 128u,
                                   (unsigned long long)K);
                tdm_load_tile_bf16(p ? bsA0 : bsA1, gB + ktn, 32u, 64u,
                                   (unsigned long long)K);
                __builtin_amdgcn_s_wait_tensorcnt(2);
            } else {
                __builtin_amdgcn_s_wait_tensorcnt(0);
            }
        }
        __syncthreads();                    // tile t visible to all waves
#else
        {
            const int kt = t << 5;
            *(uint4*)&As[p][rA][cA]      = *(const uint4*)&gA[(size_t)rA * K + kt + cA];
            *(uint4*)&As[p][64 + rA][cA] = *(const uint4*)&gA[(size_t)(64 + rA) * K + kt + cA];
            *(uint4*)&Bs[p][rA][cA]      = *(const uint4*)&gB[(size_t)rA * K + kt + cA];
        }
        __syncthreads();
#endif
        // Fragments in contiguous k order (k = 16*hi + i), same for A(16x32) and B(32x16).
        v16bf a0, a1, b0, b1;
#pragma unroll
        for (int i = 0; i < 16; ++i) {
            const int kA = i + (hi << 4);
            a0[i] = As[p][m0 + lm][kA];
            a1[i] = As[p][m0 + 16 + lm][kA];
            const int kB = i + (hi << 4);
            b0[i] = Bs[p][n0 + lm][kB];
            b1[i] = Bs[p][n0 + 16 + lm][kB];
        }
        acc00 = __builtin_amdgcn_wmma_f32_16x16x32_bf16(false, a0, false, b0,
                                                        (short)0, acc00, false, false);
        acc01 = __builtin_amdgcn_wmma_f32_16x16x32_bf16(false, a0, false, b1,
                                                        (short)0, acc01, false, false);
        acc10 = __builtin_amdgcn_wmma_f32_16x16x32_bf16(false, a1, false, b0,
                                                        (short)0, acc10, false, false);
        acc11 = __builtin_amdgcn_wmma_f32_16x16x32_bf16(false, a1, false, b1,
                                                        (short)0, acc11, false, false);
        __syncthreads();                    // buf[p] consumed; safe to overwrite
    }

    // C/D layout: VGPR v -> M = v + 8*hi (+16 per M sub-tile); N = lm (+16).
#pragma unroll
    for (int v = 0; v < 8; ++v) {
        const float rs[2][2] = {{acc00[v], acc01[v]}, {acc10[v], acc11[v]}};
#pragma unroll
        for (int ti = 0; ti < 2; ++ti) {
            const int m = mBase + m0 + ti * 16 + v + (hi << 3);
            if (m < M) {
#pragma unroll
                for (int tj = 0; tj < 2; ++tj) {
                    const int n = nBase + n0 + tj * 16 + lm;
                    float r = rs[ti][tj] + bias[n];
                    if (relu) r = fmaxf(r, 0.0f);
                    if (outF) outF[(size_t)m * N + n] = r;
                    if (outB) outB[(size_t)m * N + n] = (__bf16)r;
                }
            }
        }
    }
}

// ---------------------------------------------------------------------------
// Kernel 3: heads. One wave per (feature j, roi); 768-dot + shuffle reduce.
// Output layout (flat, reference return order):
//   [0,8000) bbox (1000,2,4) | [8000,10000) logits (1000,2)
//   [10000,14000) regress (1000,2,2): [..,0]=reg, [..,1]=unc
// ---------------------------------------------------------------------------
__global__ __launch_bounds__(32)
void heads_kernel(const float* __restrict__ x,
                  const float* __restrict__ w_bbox, const float* __restrict__ b_bbox,
                  const float* __restrict__ w_cls,  const float* __restrict__ b_cls,
                  const float* __restrict__ w_reg,  const float* __restrict__ b_reg,
                  const float* __restrict__ w_unc,  const float* __restrict__ b_unc,
                  float* __restrict__ out)
{
    const int j    = blockIdx.x;   // 0..13
    const int roi  = blockIdx.y;   // 0..999
    const int lane = threadIdx.x;  // wave32

    const float* wrow;
    float bias;
    int   outIdx;
    if (j < 8)       { wrow = w_bbox + j * DHID;        bias = b_bbox[j];
                       outIdx = roi * 8 + j; }
    else if (j < 10) { const int jj = j - 8;
                       wrow = w_cls + jj * DHID;        bias = b_cls[jj];
                       outIdx = 8000 + roi * 2 + jj; }
    else if (j < 12) { const int jj = j - 10;
                       wrow = w_reg + jj * DHID;        bias = b_reg[jj];
                       outIdx = 10000 + roi * 4 + jj * 2; }
    else             { const int jj = j - 12;
                       wrow = w_unc + jj * DHID;        bias = b_unc[jj];
                       outIdx = 10000 + roi * 4 + jj * 2 + 1; }

    const float* xr = x + (size_t)roi * DHID;
    float s = 0.0f;
    for (int k = lane; k < DHID; k += 32) s += xr[k] * wrow[k];
#pragma unroll
    for (int off = 16; off > 0; off >>= 1) s += __shfl_down(s, off, 32);
    if (lane == 0) out[outIdx] = s + bias;
}

// ---------------------------------------------------------------------------
// Host launcher
// ---------------------------------------------------------------------------
extern "C" void kernel_launch(void* const* d_in, const int* in_sizes, int n_in,
                              void* d_out, int out_size, void* d_ws, size_t ws_size,
                              hipStream_t stream)
{
    (void)in_sizes; (void)n_in; (void)out_size; (void)ws_size;

    const float* f0     = (const float*)d_in[0];
    const float* f1     = (const float*)d_in[1];
    const float* f2     = (const float*)d_in[2];
    const float* f3     = (const float*)d_in[3];
    const float* rois   = (const float*)d_in[4];
    const float* w1     = (const float*)d_in[5];   // [768,9408] fp32
    const float* b1     = (const float*)d_in[6];
    const float* w2     = (const float*)d_in[7];   // [768,768] fp32
    const float* b2     = (const float*)d_in[8];
    const float* w_bbox = (const float*)d_in[9];
    const float* b_bbox = (const float*)d_in[10];
    const float* w_cls  = (const float*)d_in[11];
    const float* b_cls  = (const float*)d_in[12];
    const float* w_reg  = (const float*)d_in[13];
    const float* b_reg  = (const float*)d_in[14];
    const float* w_unc  = (const float*)d_in[15];
    const float* b_unc  = (const float*)d_in[16];
    float* out = (float*)d_out;

    // Workspace: bf16 region first (all sub-buffers 16B aligned), then fp32.
    __bf16* wsb      = (__bf16*)d_ws;
    __bf16* pooled   = wsb;                                   // MPAD x 9408
    __bf16* w1b      = pooled + (size_t)MPAD * KFC1;          // 768 x 9408
    __bf16* w2b      = w1b    + (size_t)DHID * KFC1;          // 768 x 768
    __bf16* fc1      = w2b    + (size_t)DHID * DHID;          // MPAD x 768
    float*  fc2      = (float*)(fc1 + (size_t)MPAD * DHID);   // 1000 x 768

    // 0) weight conversion fp32 -> bf16
    {
        const int n1 = DHID * KFC1;   // 7,225,344 (mult of 1024)
        const int n2 = DHID * DHID;   //   589,824 (mult of 1024)
        cvt_f32_bf16_x4<<<n1 / 1024, 256, 0, stream>>>((const float4*)w1, w1b, n1 / 4);
        cvt_f32_bf16_x4<<<n2 / 1024, 256, 0, stream>>>((const float4*)w2, w2b, n2 / 4);
    }
    // 1) RoIAlign -> bf16 activations
    roi_align_kernel<<<NROIS, 256, 0, stream>>>(f0, f1, f2, f3, rois, pooled);

    // 2) FC1: [1024,9408] x [9408,768] -> bf16 (relu)
    {
        dim3 grid(DHID / 64, MPAD / 128);
        gemm_bf16_wmma<<<grid, 256, 0, stream>>>(pooled, w1b, b1,
                                                 nullptr, fc1,
                                                 NROIS, DHID, KFC1, 1);
    }
    // 3) FC2: [1024,768] x [768,768] -> fp32 (relu)
    {
        dim3 grid(DHID / 64, MPAD / 128);
        gemm_bf16_wmma<<<grid, 256, 0, stream>>>(fc1, w2b, b2,
                                                 fc2, nullptr,
                                                 NROIS, DHID, DHID, 1);
    }
    // 4) Heads
    {
        dim3 grid(14, NROIS);
        heads_kernel<<<grid, 32, 0, stream>>>(fc2,
                                              w_bbox, b_bbox, w_cls, b_cls,
                                              w_reg, b_reg, w_unc, b_unc, out);
    }
}
